// HeteroGNN_29815662968875
// MI455X (gfx1250) — matrix operand order repaired and checked by reading, not compiled, not run on hardware
//
#include <hip/hip_runtime.h>
#include <cstddef>

#define NA 50000
#define NP 100000
#define NE 200000
#define H  256

typedef __attribute__((ext_vector_type(16))) _Float16 v16h;
typedef __attribute__((ext_vector_type(8)))  _Float16 v8h;
typedef __attribute__((ext_vector_type(8)))  float    v8f;

// ---------------------------------------------------------------- utilities
__global__ void zero_k(float* __restrict__ p, size_t n) {
    size_t i = (size_t)blockIdx.x * blockDim.x + threadIdx.x;
    if (i < n) p[i] = 0.0f;
}

__global__ void f2h_k(const float* __restrict__ in, _Float16* __restrict__ out, size_t n) {
    size_t i = (size_t)blockIdx.x * blockDim.x + threadIdx.x;
    if (i < n) out[i] = (_Float16)in[i];
}

// degree count: one atomic per edge, computed once per direction (dst-only)
__global__ void count_k(const int* __restrict__ dst, float* __restrict__ cnt, int E) {
    int e = blockIdx.x * blockDim.x + threadIdx.x;
    if (e < E) atomicAdd(cnt + dst[e], 1.0f);
}

// edge scatter-add: one wave per edge, lanes stripe the feature dim.
// 51.2M f32 atomics per (direction,layer) -- all L2-resident on MI455X (192MB L2).
__global__ void scatter_add_k(const float* __restrict__ x, const int* __restrict__ src,
                              const int* __restrict__ dst, float* __restrict__ acc,
                              int E, int d) {
    int lane = threadIdx.x & 31;
    int e = blockIdx.x * (blockDim.x >> 5) + (threadIdx.x >> 5);
    if (e >= E) return;
    int s = src[e], t = dst[e];
    const float* xr = x + (size_t)s * d;
    float*       ar = acc + (size_t)t * d;
    for (int i = lane; i < d; i += 32) atomicAdd(ar + i, xr[i]);
}

// mean = acc / max(cnt,1), quantize to f16 (WMMA A operand)
__global__ void mean_h_k(const float* __restrict__ acc, const float* __restrict__ cnt,
                         _Float16* __restrict__ out, int M, int d) {
    size_t i = (size_t)blockIdx.x * blockDim.x + threadIdx.x;
    if (i >= (size_t)M * d) return;
    int row = (int)(i / (size_t)d);
    float c = cnt[row];
    c = c > 1.0f ? c : 1.0f;
    out[i] = (_Float16)(acc[i] / c);
}

// ------------------------------------------------- fused SAGE GEMM (WMMA f16)
// C[M,256] = A1[M,K1] @ W1[256,K1]^T + A2[M,K2] @ W2[256,K2]^T + b1 + b2
// optional ELU; writes f32 result + f16 copy for the next layer's A operand.
//
// TWO=true  (hot path): 256 threads = 8 waves; wave = (mquad, nhalf) covers a
//   32(M) x 128(N) patch = TWO 16-row tiles sharing every B fragment load.
// TWO=false (16-row remainder): wave covers 16 x 128; only waves 0 and 4 run.
// Wave id is readfirstlane'd so m0 is an SGPR value and the guard is a scalar
// branch: EXEC is never touched around WMMA (ISA requires EXEC all-1s).
template <bool TWO>
__global__ __launch_bounds__(256) void sage_gemm_k(
    const _Float16* __restrict__ A1, int K1, const _Float16* __restrict__ W1,
    const _Float16* __restrict__ A2, int K2, const _Float16* __restrict__ W2,
    const float* __restrict__ b1, const float* __restrict__ b2,
    int Mend, int mbase, float* __restrict__ Cf, _Float16* __restrict__ Ch, int do_elu) {

    const int lane  = threadIdx.x & 31;
    const int wv    = __builtin_amdgcn_readfirstlane((int)threadIdx.x) >> 5;  // scalar
    const int mquad = wv & 3;
    const int nbase = (wv >> 2) * 128;                 // 0 or 128
    const int m0 = mbase + blockIdx.x * (TWO ? 128 : 64) + mquad * (TWO ? 32 : 16);
    if (m0 >= Mend) return;                            // scalar s_cbranch: body fully skipped

    v8f acc0[8] = {};
    v8f acc1[8] = {};

    const int rsel = lane >> 4;            // 0: lanes 0-15, 1: lanes 16-31
    const int mrow = m0 + (lane & 15);
    const int koff = rsel * 16;            // A: lanes 16-31 hold K+16..31 (contiguous, matches B)

    #pragma unroll
    for (int pass = 0; pass < 2; ++pass) {
        const _Float16* A = pass ? A2 : A1;
        const _Float16* W = pass ? W2 : W1;
        const int K       = pass ? K2 : K1;
        const _Float16* arow0 = A + (size_t)mrow * K;
        const _Float16* arow1 = arow0 + (size_t)16 * K;

        for (int kb = 0; kb < K; kb += 32) {
            // A fragments: contiguous k = 16*rsel + s (two 16B contiguous loads/lane)
            v8h a0lo = *(const v8h*)(arow0 + kb + koff);
            v8h a0hi = *(const v8h*)(arow0 + kb + 8 + koff);
            v16h a0, a1;
            #pragma unroll
            for (int i = 0; i < 8; ++i) { a0[i] = a0lo[i]; a0[8 + i] = a0hi[i]; }
            if (TWO) {
                v8h a1lo = *(const v8h*)(arow1 + kb + koff);
                v8h a1hi = *(const v8h*)(arow1 + kb + 8 + koff);
                #pragma unroll
                for (int i = 0; i < 8; ++i) { a1[i] = a1lo[i]; a1[8 + i] = a1hi[i]; }
            }

            #pragma unroll
            for (int nt = 0; nt < 8; ++nt) {
                // B fragment: B[k][n] = W[n][k] -> one contiguous 32B load per lane,
                // reused by both M-tiles.
                const int ncol = nbase + nt * 16 + (lane & 15);
                v16h b = *(const v16h*)(W + (size_t)ncol * K + kb + rsel * 16);
                acc0[nt] = __builtin_amdgcn_wmma_f32_16x16x32_f16(
                    false, a0, false, b, (short)0, acc0[nt], false, false);
                if (TWO)
                    acc1[nt] = __builtin_amdgcn_wmma_f32_16x16x32_f16(
                        false, a1, false, b, (short)0, acc1[nt], false, false);
            }
        }
    }

    // epilogue: bias + ELU + f32 store + f16 requantize
    const int col0  = lane & 15;
    const int rbase = m0 + rsel * 8;       // C layout: vgpr i -> row m0 + rsel*8 + i
    #pragma unroll
    for (int nt = 0; nt < 8; ++nt) {
        const int col = nbase + nt * 16 + col0;
        const float bias = b1[col] + b2[col];
        #pragma unroll
        for (int i = 0; i < 8; ++i) {
            float v = acc0[nt][i] + bias;
            if (do_elu) v = v > 0.0f ? v : expm1f(v);
            size_t idx = (size_t)(rbase + i) * H + col;
            Cf[idx] = v;
            if (Ch) Ch[idx] = (_Float16)v;
        }
        if (TWO) {
            #pragma unroll
            for (int i = 0; i < 8; ++i) {
                float v = acc1[nt][i] + bias;
                if (do_elu) v = v > 0.0f ? v : expm1f(v);
                size_t idx = (size_t)(rbase + 16 + i) * H + col;
                Cf[idx] = v;
                if (Ch) Ch[idx] = (_Float16)v;
            }
        }
    }
}

// ------------------------------------------------------------- output head
// [50000,256] @ [10,256]^T + b : tiny (0.26 GF), plain VALU dot products.
__global__ void out_k(const float* __restrict__ x, const float* __restrict__ Wo,
                      const float* __restrict__ bo, float* __restrict__ out, int M) {
    int idx = blockIdx.x * blockDim.x + threadIdx.x;
    if (idx >= M * 10) return;
    int row = idx / 10, o = idx % 10;
    const float* xr = x + (size_t)row * H;
    const float* wr = Wo + o * H;
    float s = 0.0f;
    #pragma unroll 4
    for (int i = 0; i < H; ++i) s = fmaf(xr[i], wr[i], s);
    out[idx] = s + bo[o];
}

// --------------------------------------------------------------- orchestration
extern "C" void kernel_launch(void* const* d_in, const int* in_sizes, int n_in,
                              void* d_out, int out_size, void* d_ws, size_t ws_size,
                              hipStream_t stream) {
    const float* x_author = (const float*)d_in[0];
    const float* x_paper  = (const float*)d_in[1];
    const int*   ei_a2p   = (const int*)d_in[2];   // [2,NE]: row0=src(author) row1=dst(paper)
    const int*   ei_p2a   = (const int*)d_in[3];   // [2,NE]: row0=src(paper)  row1=dst(author)
    const float* Wl0_ap = (const float*)d_in[4];   const float* bl0_ap = (const float*)d_in[5];
    const float* Wr0_ap = (const float*)d_in[6];   const float* br0_ap = (const float*)d_in[7];
    const float* Wl0_pa = (const float*)d_in[8];   const float* bl0_pa = (const float*)d_in[9];
    const float* Wr0_pa = (const float*)d_in[10];  const float* br0_pa = (const float*)d_in[11];
    const float* Wl_ap  = (const float*)d_in[12];  const float* bl_ap  = (const float*)d_in[13];
    const float* Wr_ap  = (const float*)d_in[14];  const float* br_ap  = (const float*)d_in[15];
    const float* Wl_pa  = (const float*)d_in[16];  const float* bl_pa  = (const float*)d_in[17];
    const float* Wr_pa  = (const float*)d_in[18];  const float* br_pa  = (const float*)d_in[19];
    const float* W_out  = (const float*)d_in[20];  const float* b_out  = (const float*)d_in[21];

    // -------- workspace carve (f32 accumulators double as next-layer outputs)
    char* w = (char*)d_ws;
    auto allocf = [&](size_t n) { float* p = (float*)w; w += n * sizeof(float); return p; };
    float* pa0   = allocf((size_t)NP * H);   // paper f32 ping
    float* pa1   = allocf((size_t)NP * H);   // paper f32 pong (also scatter accum)
    float* au0   = allocf((size_t)NA * H);
    float* au1   = allocf((size_t)NA * H);
    float* cnt_p = allocf(NP);
    float* cnt_a = allocf(NA);
    auto alloch = [&](size_t n) { _Float16* p = (_Float16*)w; w += n * sizeof(_Float16); return p; };
    _Float16* ph0 = alloch((size_t)NP * H);  // paper f16 ping (starts as f16(x_paper))
    _Float16* ph1 = alloch((size_t)NP * H);
    _Float16* ah0 = alloch((size_t)NA * H);
    _Float16* ah1 = alloch((size_t)NA * H);
    _Float16* mp  = alloch((size_t)NP * H);  // mean into paper (f16)
    _Float16* ma  = alloch((size_t)NA * H);  // mean into author (f16)
    _Float16* xah = alloch((size_t)NA * 128);
    _Float16* h_wl0ap = alloch(H * 128);
    _Float16* h_wr0ap = alloch(H * H);
    _Float16* h_wl0pa = alloch(H * H);
    _Float16* h_wr0pa = alloch(H * 128);
    _Float16* h_wlap  = alloch(2 * H * H);
    _Float16* h_wrap  = alloch(2 * H * H);
    _Float16* h_wlpa  = alloch(2 * H * H);
    _Float16* h_wrpa  = alloch(2 * H * H);

    auto zero = [&](float* p, size_t n) {
        zero_k<<<(unsigned)((n + 255) / 256), 256, 0, stream>>>(p, n);
    };
    auto f2h = [&](const float* in, _Float16* out, size_t n) {
        f2h_k<<<(unsigned)((n + 255) / 256), 256, 0, stream>>>(in, out, n);
    };
    auto scat = [&](const float* x, const int* s, const int* d, float* acc, int dim) {
        scatter_add_k<<<(NE + 7) / 8, 256, 0, stream>>>(x, s, d, acc, NE, dim);
    };
    auto mean = [&](const float* acc, const float* cnt, _Float16* out, int M, int dim) {
        mean_h_k<<<(unsigned)(((size_t)M * dim + 255) / 256), 256, 0, stream>>>(acc, cnt, out, M, dim);
    };
    auto gemm = [&](const _Float16* A1, int K1, const _Float16* W1,
                    const _Float16* A2, int K2, const _Float16* W2,
                    const float* b1, const float* b2, int M,
                    float* Cf, _Float16* Ch, int elu) {
        const int Mfull = (M / 32) * 32;                 // pair-tile region
        if (Mfull > 0) {
            int blocks = (Mfull + 127) / 128;            // 8 waves/block, 32x128 per wave
            sage_gemm_k<true><<<blocks, 256, 0, stream>>>(A1, K1, W1, A2, K2, W2,
                                                          b1, b2, Mfull, 0, Cf, Ch, elu);
        }
        if (M > Mfull) {                                 // single 16-row remainder tile
            sage_gemm_k<false><<<1, 256, 0, stream>>>(A1, K1, W1, A2, K2, W2,
                                                      b1, b2, M, Mfull, Cf, Ch, elu);
        }
    };

    // -------- weight + feature quantization (f32 -> f16)
    f2h(Wl0_ap, h_wl0ap, H * 128);   f2h(Wr0_ap, h_wr0ap, H * H);
    f2h(Wl0_pa, h_wl0pa, H * H);     f2h(Wr0_pa, h_wr0pa, H * 128);
    f2h(Wl_ap, h_wlap, 2 * H * H);   f2h(Wr_ap, h_wrap, 2 * H * H);
    f2h(Wl_pa, h_wlpa, 2 * H * H);   f2h(Wr_pa, h_wrpa, 2 * H * H);
    f2h(x_author, xah, (size_t)NA * 128);
    f2h(x_paper,  ph0, (size_t)NP * H);

    // -------- degree counts (dst-only -> once per direction)
    zero(cnt_p, NP);  zero(cnt_a, NA);
    count_k<<<(NE + 255) / 256, 256, 0, stream>>>(ei_a2p + NE, cnt_p, NE);
    count_k<<<(NE + 255) / 256, 256, 0, stream>>>(ei_p2a + NE, cnt_a, NE);

    // -------- layer 0 (lazy in-dims: author 128, paper 256)
    zero(pa1, (size_t)NP * 128);
    scat(x_author, ei_a2p, ei_a2p + NE, pa1, 128);
    mean(pa1, cnt_p, mp, NP, 128);
    zero(au1, (size_t)NA * H);
    scat(x_paper, ei_p2a, ei_p2a + NE, au1, H);
    mean(au1, cnt_a, ma, NA, H);
    gemm(mp, 128, h_wl0ap, ph0, H, h_wr0ap, bl0_ap, br0_ap, NP, pa0, ph1, 1);
    gemm(ma, H,   h_wl0pa, xah, 128, h_wr0pa, bl0_pa, br0_pa, NA, au0, ah0, 1);

    // -------- layers 1..2 (hidden 256, fused K = 512 per GEMM)
    float *pcur = pa0, *pfree = pa1, *acur = au0, *afree = au1;
    _Float16 *phc = ph1, *phf = ph0, *ahc = ah0, *ahf = ah1;
    for (int l = 0; l < 2; ++l) {
        zero(pfree, (size_t)NP * H);
        scat(acur, ei_a2p, ei_a2p + NE, pfree, H);
        mean(pfree, cnt_p, mp, NP, H);
        zero(afree, (size_t)NA * H);
        scat(pcur, ei_p2a, ei_p2a + NE, afree, H);
        mean(afree, cnt_a, ma, NA, H);
        gemm(mp, H, h_wlap + (size_t)l * H * H, phc, H, h_wrap + (size_t)l * H * H,
             bl_ap + l * H, br_ap + l * H, NP, pfree, phf, 1);
        gemm(ma, H, h_wlpa + (size_t)l * H * H, ahc, H, h_wrpa + (size_t)l * H * H,
             bl_pa + l * H, br_pa + l * H, NA, afree, ahf, 1);
        float* t;      t = pcur; pcur = pfree; pfree = t;
                       t = acur; acur = afree; afree = t;
        _Float16* th;  th = phc; phc = phf; phf = th;
                       th = ahc; ahc = ahf; ahf = th;
    }

    // -------- output head
    out_k<<<(NA * 10 + 255) / 256, 256, 0, stream>>>(acur, W_out, b_out, (float*)d_out, NA);
}
